// LongformerSummaryGenerator_57475252355351
// MI455X (gfx1250) — matrix operand order repaired and checked by reading, not compiled, not run on hardware
//
#include <hip/hip_runtime.h>
#include <math.h>

#define LAYERS 4
#define HEADS  12
#define DHEAD  64
#define HIDN   768
#define FFND   3072
#define VOCAB  50265
#define SEQ    4096
#define WIN    256
#define NCHUNK 16          // SEQ / WIN
#define NGLOB  16
#define BANDW  784         // NGLOB + 3*WIN
#define BANDP  800         // BANDW padded to multiple of 32
#define NEGF   (-1.0e9f)

#define GLOBAL_AS __attribute__((address_space(1)))

typedef unsigned short u16;
typedef __attribute__((ext_vector_type(16))) __bf16 v16bf;
typedef __attribute__((ext_vector_type(8)))  __bf16 v8bf;
typedef __attribute__((ext_vector_type(8)))  float  v8f;

// ---------------- scalar helpers ----------------

__device__ inline u16 f2bf(float f) {
    unsigned u = __builtin_bit_cast(unsigned, f);
    u += 0x7FFFu + ((u >> 16) & 1u);        // round-to-nearest-even
    return (u16)(u >> 16);
}
__device__ inline float bf2f(u16 h) {
    unsigned u = ((unsigned)h) << 16;
    return __builtin_bit_cast(float, u);
}
__device__ inline float gelu_f(float x) {
    return 0.5f * x * (1.0f + erff(x * 0.70710678118654752f));
}
__device__ inline v8f zero_v8f() {
    v8f z = {0.f, 0.f, 0.f, 0.f, 0.f, 0.f, 0.f, 0.f};
    return z;
}
__device__ inline int iclamp(int v, int lo, int hi) {
    return v < lo ? lo : (v > hi ? hi : v);
}

__device__ inline v8f wmma_bf16(v16bf a, v16bf b, v8f c) {
    return __builtin_amdgcn_wmma_f32_16x16x32_bf16(false, a, false, b, (short)0, c,
                                                   false, false);
}

// Explicit global-address-space stores: guarantees global_store encodings
// (STOREcnt only) instead of flat_store (STOREcnt + DScnt + aperture check).
__device__ inline void gst_f32(float* p, float v) {
    *(GLOBAL_AS float*)(unsigned long long)p = v;
}
__device__ inline void gst_u16(u16* p, u16 v) {
    *(GLOBAL_AS u16*)(unsigned long long)p = v;
}

// Assemble a 16-element bf16 fragment from two contiguous 8-element chunks at
// p and p+hioff (A and B frags: hioff=8). Always unconditional:
// callers clamp row indices so loads stay in-bounds; garbage values only ever
// land in output elements that are masked or never stored. Loads are forced
// into the global address space -> global_load_b128 (LOADcnt only, no DScnt).
__device__ inline v16bf frag_ld(const u16* __restrict__ p, int hioff) {
    const GLOBAL_AS u16* g = (const GLOBAL_AS u16*)(unsigned long long)p;
    v8bf lo = *(const GLOBAL_AS v8bf*)g;
    v8bf hi = *(const GLOBAL_AS v8bf*)(g + hioff);
    v16bf f;
#pragma unroll
    for (int i = 0; i < 8; ++i) { f[i] = lo[i]; f[8 + i] = hi[i]; }
    return f;
}

// ---------------- generic bf16 WMMA GEMM ----------------
// C[M,N] = A[M,K](bf16) * Bt[N,K]^T(bf16) + bias ; optional exact-GELU epilogue.
// Block = 128 threads = 4 waves in a 2x2 arrangement; each wave owns a 64x32
// macro tile (4x2 WMMA subtiles) -> block tile is 128x64. Single unguarded
// load path (rows/cols clamped); only the stores are bounds-guarded.
__global__ __launch_bounds__(128) void gemm_bf16_kernel(
    const u16* __restrict__ A, int lda, const u16* __restrict__ Bt, int ldb,
    const float* __restrict__ bias, float* __restrict__ Cf, u16* __restrict__ Cb,
    int ldc, int M, int N, int K, int act) {
    int wave = (int)(threadIdx.x >> 5);
    int lane = (int)(threadIdx.x & 31);
    int half = lane >> 4;
    int ml = lane & 15;
    int nl = lane & 15;
    int m0 = blockIdx.y * 128 + (wave >> 1) * 64;
    int n0 = blockIdx.x * 64 + (wave & 1) * 32;

    v8f acc[4][2];
#pragma unroll
    for (int i = 0; i < 4; ++i)
#pragma unroll
        for (int j = 0; j < 2; ++j) acc[i][j] = zero_v8f();

    const u16* pa[4];
    const u16* pb[2];
#pragma unroll
    for (int i = 0; i < 4; ++i) {
        int row = iclamp(m0 + 16 * i + ml, 0, M - 1);
        pa[i] = A + (size_t)row * lda + 16 * half;
    }
#pragma unroll
    for (int j = 0; j < 2; ++j) {
        int col = iclamp(n0 + 16 * j + nl, 0, N - 1);
        pb[j] = Bt + (size_t)col * ldb + 16 * half;
    }

    for (int k0 = 0; k0 < K; k0 += 32) {
        v16bf a0 = frag_ld(pa[0], 8);
        v16bf a1 = frag_ld(pa[1], 8);
        v16bf a2 = frag_ld(pa[2], 8);
        v16bf a3 = frag_ld(pa[3], 8);
        v16bf b0 = frag_ld(pb[0], 8);
        v16bf b1 = frag_ld(pb[1], 8);
        // gfx1250 prefetch: pull next K panels toward the WGP.
        __builtin_prefetch((const void*)(pa[0] + 512), 0, 1);
        __builtin_prefetch((const void*)(pb[0] + 512), 0, 1);
        acc[0][0] = wmma_bf16(a0, b0, acc[0][0]);
        acc[0][1] = wmma_bf16(a0, b1, acc[0][1]);
        acc[1][0] = wmma_bf16(a1, b0, acc[1][0]);
        acc[1][1] = wmma_bf16(a1, b1, acc[1][1]);
        acc[2][0] = wmma_bf16(a2, b0, acc[2][0]);
        acc[2][1] = wmma_bf16(a2, b1, acc[2][1]);
        acc[3][0] = wmma_bf16(a3, b0, acc[3][0]);
        acc[3][1] = wmma_bf16(a3, b1, acc[3][1]);
#pragma unroll
        for (int i = 0; i < 4; ++i) pa[i] += 32;
#pragma unroll
        for (int j = 0; j < 2; ++j) pb[j] += 32;
    }

#pragma unroll
    for (int i = 0; i < 4; ++i) {
#pragma unroll
        for (int j = 0; j < 2; ++j) {
            int nb = n0 + 16 * j + nl;
            if (nb >= N) continue;
            float bv = bias ? bias[nb] : 0.0f;
#pragma unroll
            for (int r = 0; r < 8; ++r) {
                int m = m0 + 16 * i + r + 8 * half;   // C layout: VGPR r -> M=r(+8)
                if (m < M) {
                    float v = acc[i][j][r] + bv;
                    if (act == 1) v = gelu_f(v);
                    if (Cf) gst_f32(Cf + (size_t)m * ldc + nb, v);
                    if (Cb) gst_u16(Cb + (size_t)m * ldc + nb, f2bf(v));
                }
            }
        }
    }
}

// ---------------- band attention scores ----------------
// scores[h][c][m 0..255][n 0..783]; n<16: global keys, n>=16: band key j=n-16
// (key row = c*W - W + j). Invalid rows are loaded clamped and then overwritten
// with NEG by the epilogue mask (which uses the unclamped row index).
__global__ __launch_bounds__(128) void band_score_kernel(
    const u16* __restrict__ QH, const u16* __restrict__ KH,
    const int* __restrict__ amask, const int* __restrict__ gmask,
    u16* __restrict__ SB) {
    int hc = blockIdx.z;
    int h = hc / NCHUNK, c = hc % NCHUNK;
    int wave = (int)(threadIdx.x >> 5);
    int lane = (int)(threadIdx.x & 31);
    int half = lane >> 4, nl = lane & 15;
    int ntile = blockIdx.x * 4 + wave;
    if (ntile >= 49) return;
    int n0 = ntile * 16;
    int m0 = blockIdx.y * 16;
    const u16* Qh = QH + (size_t)h * SEQ * DHEAD;
    const u16* Kh = KH + (size_t)h * SEQ * DHEAD;
    int n = n0 + nl;
    int srow = (n < NGLOB) ? n : (c * WIN - WIN + (n - NGLOB));
    int srow_cl = iclamp(srow, 0, SEQ - 1);
    v8f acc = zero_v8f();
#pragma unroll
    for (int k0 = 0; k0 < DHEAD; k0 += 32) {
        v16bf a = frag_ld(Qh + (size_t)(c * WIN + m0 + nl) * DHEAD + k0 + 16 * half, 8);
        v16bf b = frag_ld(Kh + (size_t)srow_cl * DHEAD + k0 + 16 * half, 8);
        acc = wmma_bf16(a, b, acc);
    }
#pragma unroll
    for (int r = 0; r < 8; ++r) {
        int m = m0 + r + 8 * half;          // query pos within chunk
        bool ok;
        if (n < NGLOB) {
            ok = amask[n] > 0;
        } else {
            int j = n - NGLOB;
            int rel = j - m;                 // kpos - qpos + WIN
            ok = (rel >= 0) && (rel <= 2 * WIN) && (srow >= 0) && (srow < SEQ) &&
                 (amask[srow_cl] > 0) && (gmask[srow_cl] == 0);
        }
        gst_u16(SB + ((size_t)hc * WIN + m) * BANDP + n, f2bf(ok ? acc[r] : NEGF));
    }
}

// ---------------- band attention output ----------------
// out[256x64] = probs[256x800] x Vcat[800x64]; Vcat row k<16 -> v global row k,
// 16<=k<784 -> v row c*W-W+(k-16). Invalid/pad K positions have probs == +0.0
// exactly (softmax of NEG underflows), so clamped (finite) V rows contribute 0.
__global__ __launch_bounds__(128) void attn_out_kernel(const u16* __restrict__ SB,
                                                       const u16* __restrict__ VT,
                                                       float* __restrict__ ATTF) {
    int hc = blockIdx.z;
    int h = hc / NCHUNK, c = hc % NCHUNK;
    int wave = (int)(threadIdx.x >> 5);
    int lane = (int)(threadIdx.x & 31);
    int half = lane >> 4, nl = lane & 15;
    int n0 = wave * 16;                     // d within head
    int m0 = blockIdx.y * 16;
    const u16* P = SB + (size_t)hc * WIN * BANDP;
    const u16* Vh = VT + (size_t)h * DHEAD * SEQ;
    int d = n0 + nl;
    v8f acc = zero_v8f();
    for (int k0 = 0; k0 < BANDP; k0 += 32) {
        v16bf a = frag_ld(P + (size_t)(m0 + nl) * BANDP + k0 + 16 * half, 8);
        int kb = k0 + 16 * half;
        int s0 = (kb < NGLOB) ? kb : (c * WIN - WIN + (kb - NGLOB));
        s0 = iclamp(s0, 0, SEQ - 16);
        v16bf b = frag_ld(Vh + (size_t)d * SEQ + s0, 8);
        acc = wmma_bf16(a, b, acc);
    }
#pragma unroll
    for (int r = 0; r < 8; ++r) {
        int m = m0 + r + 8 * half;
        gst_f32(ATTF + (size_t)(c * WIN + m) * HIDN + h * DHEAD + d, acc[r]);
    }
}

// ---------------- global attention scores (16 x 4096 per head) ----------------
__global__ __launch_bounds__(128) void gscore_kernel(const u16* __restrict__ QGH,
                                                     const u16* __restrict__ KGH,
                                                     const int* __restrict__ amask,
                                                     u16* __restrict__ SG) {
    int h = blockIdx.z;
    int wave = (int)(threadIdx.x >> 5);
    int lane = (int)(threadIdx.x & 31);
    int half = lane >> 4, nl = lane & 15;
    int n0 = (blockIdx.x * 4 + wave) * 16;
    const u16* Qh = QGH + (size_t)h * NGLOB * DHEAD;
    const u16* Kh = KGH + (size_t)h * SEQ * DHEAD;
    v8f acc = zero_v8f();
#pragma unroll
    for (int k0 = 0; k0 < DHEAD; k0 += 32) {
        v16bf a = frag_ld(Qh + (size_t)nl * DHEAD + k0 + 16 * half, 8);
        v16bf b = frag_ld(Kh + (size_t)(n0 + nl) * DHEAD + k0 + 16 * half, 8);
        acc = wmma_bf16(a, b, acc);
    }
    int n = n0 + nl;
    bool ok = amask[n] > 0;
#pragma unroll
    for (int r = 0; r < 8; ++r) {
        int m = r + 8 * half;
        gst_u16(SG + ((size_t)h * NGLOB + m) * SEQ + n, f2bf(ok ? acc[r] : NEGF));
    }
}

// ---------------- global attention output (16 x 64 per head) ----------------
__global__ __launch_bounds__(128) void gout_kernel(const u16* __restrict__ SG,
                                                   const u16* __restrict__ VGT,
                                                   float* __restrict__ ATTF) {
    int h = blockIdx.z;
    int wave = (int)(threadIdx.x >> 5);
    int lane = (int)(threadIdx.x & 31);
    int half = lane >> 4, nl = lane & 15;
    int n0 = wave * 16;
    const u16* P = SG + (size_t)h * NGLOB * SEQ;
    const u16* Vh = VGT + (size_t)h * DHEAD * SEQ;
    int d = n0 + nl;
    v8f acc = zero_v8f();
    for (int k0 = 0; k0 < SEQ; k0 += 32) {
        v16bf a = frag_ld(P + (size_t)nl * SEQ + k0 + 16 * half, 8);
        v16bf b = frag_ld(Vh + (size_t)d * SEQ + k0 + 16 * half, 8);
        acc = wmma_bf16(a, b, acc);
    }
#pragma unroll
    for (int r = 0; r < 8; ++r) {
        int m = r + 8 * half;
        gst_f32(ATTF + (size_t)m * HIDN + h * DHEAD + d, acc[r]);  // rows < NGLOB
    }
}

// ---------------- row softmax over bf16 (in place), zero the tail pad ----------------
__global__ __launch_bounds__(128) void softmax_kernel(u16* __restrict__ buf, int width,
                                                      int stride) {
    __shared__ float red[128];
    int tid = (int)threadIdx.x;
    u16* p = buf + (size_t)blockIdx.x * stride;
    float mx = -3.4e38f;
    for (int i = tid; i < width; i += 128) mx = fmaxf(mx, bf2f(p[i]));
    red[tid] = mx; __syncthreads();
    for (int off = 64; off > 0; off >>= 1) {
        if (tid < off) red[tid] = fmaxf(red[tid], red[tid + off]);
        __syncthreads();
    }
    float smax = red[0]; __syncthreads();
    float sum = 0.f;
    for (int i = tid; i < width; i += 128) sum += __expf(bf2f(p[i]) - smax);
    red[tid] = sum; __syncthreads();
    for (int off = 64; off > 0; off >>= 1) {
        if (tid < off) red[tid] += red[tid + off];
        __syncthreads();
    }
    float inv = 1.0f / red[0];
    for (int i = tid; i < width; i += 128)
        p[i] = f2bf(__expf(bf2f(p[i]) - smax) * inv);
    for (int i = width + tid; i < stride; i += 128) p[i] = 0;
}

// ---------------- embeddings + layernorm ----------------
__global__ __launch_bounds__(256) void embed_ln_kernel(
    const int* __restrict__ ids, const float* __restrict__ wemb,
    const float* __restrict__ pemb, const float* __restrict__ temb,
    const float* __restrict__ sc, const float* __restrict__ bi,
    float* __restrict__ XF, u16* __restrict__ XB) {
    __shared__ float red[256];
    int t = blockIdx.x, tid = (int)threadIdx.x;
    int id = ids[t];
    float v[3];
    float s = 0.f;
#pragma unroll
    for (int j = 0; j < 3; ++j) {
        int idx = tid + j * 256;
        v[j] = wemb[(size_t)id * HIDN + idx] + pemb[(size_t)(t + 2) * HIDN + idx] +
               temb[idx];
        s += v[j];
    }
    red[tid] = s; __syncthreads();
    for (int off = 128; off > 0; off >>= 1) {
        if (tid < off) red[tid] += red[tid + off];
        __syncthreads();
    }
    float mean = red[0] * (1.0f / HIDN); __syncthreads();
    s = 0.f;
#pragma unroll
    for (int j = 0; j < 3; ++j) { float d = v[j] - mean; s += d * d; }
    red[tid] = s; __syncthreads();
    for (int off = 128; off > 0; off >>= 1) {
        if (tid < off) red[tid] += red[tid + off];
        __syncthreads();
    }
    float rstd = rsqrtf(red[0] * (1.0f / HIDN) + 1e-5f);
#pragma unroll
    for (int j = 0; j < 3; ++j) {
        int idx = tid + j * 256;
        float o = (v[j] - mean) * rstd * sc[idx] + bi[idx];
        XF[(size_t)t * HIDN + idx] = o;
        XB[(size_t)t * HIDN + idx] = f2bf(o);
    }
}

// ---------------- residual + layernorm: X = LN(X + Y) ----------------
__global__ __launch_bounds__(256) void resid_ln_kernel(
    float* __restrict__ X, const float* __restrict__ Y, const float* __restrict__ sc,
    const float* __restrict__ bi, u16* __restrict__ XB) {
    __shared__ float red[256];
    int t = blockIdx.x, tid = (int)threadIdx.x;
    float v[3];
    float s = 0.f;
#pragma unroll
    for (int j = 0; j < 3; ++j) {
        int idx = tid + j * 256;
        v[j] = X[(size_t)t * HIDN + idx] + Y[(size_t)t * HIDN + idx];
        s += v[j];
    }
    red[tid] = s; __syncthreads();
    for (int off = 128; off > 0; off >>= 1) {
        if (tid < off) red[tid] += red[tid + off];
        __syncthreads();
    }
    float mean = red[0] * (1.0f / HIDN); __syncthreads();
    s = 0.f;
#pragma unroll
    for (int j = 0; j < 3; ++j) { float d = v[j] - mean; s += d * d; }
    red[tid] = s; __syncthreads();
    for (int off = 128; off > 0; off >>= 1) {
        if (tid < off) red[tid] += red[tid + off];
        __syncthreads();
    }
    float rstd = rsqrtf(red[0] * (1.0f / HIDN) + 1e-5f);
#pragma unroll
    for (int j = 0; j < 3; ++j) {
        int idx = tid + j * 256;
        float o = (v[j] - mean) * rstd * sc[idx] + bi[idx];
        X[(size_t)t * HIDN + idx] = o;
        XB[(size_t)t * HIDN + idx] = f2bf(o);
    }
}

// ---------------- head split / transpose helpers ----------------
__global__ void split_qkv_kernel(const float* __restrict__ QF,
                                 const float* __restrict__ KF,
                                 const float* __restrict__ VF, u16* __restrict__ QH,
                                 u16* __restrict__ KH, u16* __restrict__ VT) {
    size_t i = (size_t)blockIdx.x * blockDim.x + threadIdx.x;
    if (i >= (size_t)SEQ * HIDN) return;
    int s = (int)(i / HIDN), col = (int)(i % HIDN);
    int h = col >> 6, d = col & 63;
    QH[((size_t)h * SEQ + s) * DHEAD + d] = f2bf(QF[i] * 0.125f);  // 1/sqrt(64)
    KH[((size_t)h * SEQ + s) * DHEAD + d] = f2bf(KF[i]);
    VT[((size_t)h * DHEAD + d) * SEQ + s] = f2bf(VF[i]);
}

__global__ void split_global_kernel(const float* __restrict__ KF,
                                    const float* __restrict__ VF,
                                    u16* __restrict__ KGH, u16* __restrict__ VGT) {
    size_t i = (size_t)blockIdx.x * blockDim.x + threadIdx.x;
    if (i >= (size_t)SEQ * HIDN) return;
    int s = (int)(i / HIDN), col = (int)(i % HIDN);
    int h = col >> 6, d = col & 63;
    KGH[((size_t)h * SEQ + s) * DHEAD + d] = f2bf(KF[i]);
    VGT[((size_t)h * DHEAD + d) * SEQ + s] = f2bf(VF[i]);
}

__global__ void split_qg_kernel(const float* __restrict__ QGF, u16* __restrict__ QGH) {
    int i = (int)(blockIdx.x * blockDim.x + threadIdx.x);
    if (i >= NGLOB * HIDN) return;
    int m = i / HIDN, col = i % HIDN;
    int h = col >> 6, d = col & 63;
    QGH[((size_t)h * NGLOB + m) * DHEAD + d] = f2bf(QGF[i] * 0.125f);
}

__global__ void f32_to_bf16_kernel(const float* __restrict__ in, u16* __restrict__ out,
                                   size_t n) {
    size_t i = (size_t)blockIdx.x * blockDim.x + threadIdx.x;
    if (i < n) out[i] = f2bf(in[i]);
}

// in f32 [K][N] -> out bf16 [N][K]
__global__ __launch_bounds__(256) void transpose_f32_bf16_kernel(
    const float* __restrict__ in, u16* __restrict__ out, int K, int N) {
    __shared__ float tile[32][33];
    int x = blockIdx.x * 32 + threadIdx.x;   // N index
    int y0 = blockIdx.y * 32;
    for (int j = (int)threadIdx.y; j < 32; j += 8) {
        int k = y0 + j;
        tile[j][threadIdx.x] = (k < K && x < N) ? in[(size_t)k * N + x] : 0.f;
    }
    __syncthreads();
    int k2 = y0 + (int)threadIdx.x;          // K index for output
    int nb = blockIdx.x * 32;
    for (int j = (int)threadIdx.y; j < 32; j += 8) {
        int n = nb + j;
        if (n < N && k2 < K) out[(size_t)n * K + k2] = f2bf(tile[threadIdx.x][j]);
    }
}

// ---------------- host orchestration ----------------
extern "C" void kernel_launch(void* const* d_in, const int* in_sizes, int n_in,
                              void* d_out, int out_size, void* d_ws, size_t ws_size,
                              hipStream_t stream) {
    (void)in_sizes; (void)n_in; (void)out_size; (void)ws_size;
    const int*   ids   = (const int*)d_in[0];
    const int*   amask = (const int*)d_in[1];
    const int*   gmask = (const int*)d_in[2];
    const float* wemb  = (const float*)d_in[3];
    const float* pemb  = (const float*)d_in[4];
    const float* temb  = (const float*)d_in[5];
    const float* eln_s = (const float*)d_in[6];
    const float* eln_b = (const float*)d_in[7];
    const float* Wfc   = (const float*)d_in[8];
    const float* bfc   = (const float*)d_in[9];
    const float* W1    = (const float*)d_in[10];
    const float* b1    = (const float*)d_in[11];
    const float* W2    = (const float*)d_in[12];
    const float* b2    = (const float*)d_in[13];
    const float* Wsrc[7] = {(const float*)d_in[14], (const float*)d_in[15],
                            (const float*)d_in[16], (const float*)d_in[17],
                            (const float*)d_in[18], (const float*)d_in[19],
                            (const float*)d_in[20]};
    const float* bq  = (const float*)d_in[21];
    const float* bk  = (const float*)d_in[22];
    const float* bv  = (const float*)d_in[23];
    const float* bo  = (const float*)d_in[24];
    const float* bqg = (const float*)d_in[25];
    const float* bkg = (const float*)d_in[26];
    const float* bvg = (const float*)d_in[27];
    const float* ln1_s = (const float*)d_in[28];
    const float* ln2_s = (const float*)d_in[29];
    const float* ln1_b = (const float*)d_in[30];
    const float* ln2_b = (const float*)d_in[31];

    char* ws = (char*)d_ws;
    size_t off = 0;
    auto alloc = [&](size_t bytes) -> char* {
        off = (off + 255) & ~(size_t)255;
        char* p = ws + off;
        off += bytes;
        return p;
    };

    float* XF   = (float*)alloc((size_t)SEQ * HIDN * 4);
    u16*   XB   = (u16*)alloc((size_t)SEQ * HIDN * 2);
    float* QF   = (float*)alloc((size_t)SEQ * HIDN * 4);
    float* KF   = (float*)alloc((size_t)SEQ * HIDN * 4);
    float* VF   = (float*)alloc((size_t)SEQ * HIDN * 4);
    float* QGF  = (float*)alloc((size_t)NGLOB * HIDN * 4);
    u16*   QH   = (u16*)alloc((size_t)SEQ * HIDN * 2);
    u16*   KH   = (u16*)alloc((size_t)SEQ * HIDN * 2);
    u16*   VTb  = (u16*)alloc((size_t)SEQ * HIDN * 2);
    u16*   KGH  = (u16*)alloc((size_t)SEQ * HIDN * 2);
    u16*   VGT  = (u16*)alloc((size_t)SEQ * HIDN * 2);
    u16*   QGH  = (u16*)alloc((size_t)HEADS * NGLOB * DHEAD * 2);
    u16*   SB   = (u16*)alloc((size_t)HEADS * NCHUNK * WIN * BANDP * 2);
    u16*   SG   = (u16*)alloc((size_t)HEADS * NGLOB * SEQ * 2);
    float* ATTF = (float*)alloc((size_t)SEQ * HIDN * 4);
    u16*   ATTB = (u16*)alloc((size_t)SEQ * HIDN * 2);
    float* OF   = (float*)alloc((size_t)SEQ * HIDN * 4);
    u16*   HB   = (u16*)alloc((size_t)SEQ * FFND * 2);
    u16*   WT[7];
    for (int i = 0; i < 7; ++i) WT[i] = (u16*)alloc((size_t)LAYERS * HIDN * HIDN * 2);
    u16* W1T  = (u16*)alloc((size_t)LAYERS * FFND * HIDN * 2);  // per layer [FFN][HID]
    u16* W2T  = (u16*)alloc((size_t)LAYERS * HIDN * FFND * 2);  // per layer [HID][FFN]
    u16* WfcT = (u16*)alloc((size_t)VOCAB * HIDN * 2);          // [V][HID]

    // ---- weight convert + transpose (bf16, Bt layout) ----
    dim3 tb(32, 8);
    for (int l = 0; l < LAYERS; ++l) {
        for (int i = 0; i < 7; ++i) {
            transpose_f32_bf16_kernel<<<dim3(HIDN / 32, HIDN / 32), tb, 0, stream>>>(
                Wsrc[i] + (size_t)l * HIDN * HIDN, WT[i] + (size_t)l * HIDN * HIDN,
                HIDN, HIDN);
        }
        transpose_f32_bf16_kernel<<<dim3(FFND / 32, HIDN / 32), tb, 0, stream>>>(
            W1 + (size_t)l * HIDN * FFND, W1T + (size_t)l * FFND * HIDN, HIDN, FFND);
        transpose_f32_bf16_kernel<<<dim3(HIDN / 32, FFND / 32), tb, 0, stream>>>(
            W2 + (size_t)l * FFND * HIDN, W2T + (size_t)l * HIDN * FFND, FFND, HIDN);
    }
    transpose_f32_bf16_kernel<<<dim3((VOCAB + 31) / 32, HIDN / 32), tb, 0, stream>>>(
        Wfc, WfcT, HIDN, VOCAB);

    // ---- embeddings ----
    embed_ln_kernel<<<SEQ, 256, 0, stream>>>(ids, wemb, pemb, temb, eln_s, eln_b, XF,
                                             XB);

    auto gemm = [&](const u16* A, int lda, const u16* Bt, int ldb, const float* bias,
                    float* Cf, u16* Cb, int ldc, int M, int N, int K, int act) {
        dim3 grid((N + 63) / 64, (M + 127) / 128, 1);
        gemm_bf16_kernel<<<grid, 128, 0, stream>>>(A, lda, Bt, ldb, bias, Cf, Cb, ldc,
                                                   M, N, K, act);
    };

    const size_t nSH = (size_t)SEQ * HIDN;
    for (int l = 0; l < LAYERS; ++l) {
        const u16* WqT  = WT[0] + (size_t)l * HIDN * HIDN;
        const u16* WkT  = WT[1] + (size_t)l * HIDN * HIDN;
        const u16* WvT  = WT[2] + (size_t)l * HIDN * HIDN;
        const u16* WoT  = WT[3] + (size_t)l * HIDN * HIDN;
        const u16* WqgT = WT[4] + (size_t)l * HIDN * HIDN;
        const u16* WkgT = WT[5] + (size_t)l * HIDN * HIDN;
        const u16* WvgT = WT[6] + (size_t)l * HIDN * HIDN;

        // q,k,v projections
        gemm(XB, HIDN, WqT, HIDN, bq + l * HIDN, QF, nullptr, HIDN, SEQ, HIDN, HIDN, 0);
        gemm(XB, HIDN, WkT, HIDN, bk + l * HIDN, KF, nullptr, HIDN, SEQ, HIDN, HIDN, 0);
        gemm(XB, HIDN, WvT, HIDN, bv + l * HIDN, VF, nullptr, HIDN, SEQ, HIDN, HIDN, 0);
        split_qkv_kernel<<<(unsigned)((nSH + 255) / 256), 256, 0, stream>>>(QF, KF, VF,
                                                                            QH, KH, VTb);
        // banded local attention
        band_score_kernel<<<dim3(13, 16, HEADS * NCHUNK), 128, 0, stream>>>(QH, KH,
                                                                            amask, gmask,
                                                                            SB);
        softmax_kernel<<<HEADS * NCHUNK * WIN, 128, 0, stream>>>(SB, BANDW, BANDP);
        attn_out_kernel<<<dim3(1, 16, HEADS * NCHUNK), 128, 0, stream>>>(SB, VTb, ATTF);

        // global attention (first NGLOB tokens attend to everything)
        gemm(XB, HIDN, WqgT, HIDN, bqg + l * HIDN, QGF, nullptr, HIDN, NGLOB, HIDN,
             HIDN, 0);
        gemm(XB, HIDN, WkgT, HIDN, bkg + l * HIDN, KF, nullptr, HIDN, SEQ, HIDN, HIDN,
             0);
        gemm(XB, HIDN, WvgT, HIDN, bvg + l * HIDN, VF, nullptr, HIDN, SEQ, HIDN, HIDN,
             0);
        split_global_kernel<<<(unsigned)((nSH + 255) / 256), 256, 0, stream>>>(KF, VF,
                                                                               KGH, VGT);
        split_qg_kernel<<<(NGLOB * HIDN + 255) / 256, 256, 0, stream>>>(QGF, QGH);
        gscore_kernel<<<dim3(SEQ / 64, 1, HEADS), 128, 0, stream>>>(QGH, KGH, amask, SG);
        softmax_kernel<<<HEADS * NGLOB, 128, 0, stream>>>(SG, SEQ, SEQ);
        gout_kernel<<<dim3(1, 1, HEADS), 128, 0, stream>>>(SG, VGT, ATTF);

        // output projection + residual LN
        f32_to_bf16_kernel<<<(unsigned)((nSH + 255) / 256), 256, 0, stream>>>(ATTF, ATTB,
                                                                              nSH);
        gemm(ATTB, HIDN, WoT, HIDN, bo + l * HIDN, OF, nullptr, HIDN, SEQ, HIDN, HIDN,
             0);
        resid_ln_kernel<<<SEQ, 256, 0, stream>>>(XF, OF, ln1_s + l * HIDN,
                                                 ln1_b + l * HIDN, XB);
        // FFN (GELU fused into GEMM epilogue, bf16 out for the next GEMM)
        gemm(XB, HIDN, W1T + (size_t)l * FFND * HIDN, HIDN, b1 + l * FFND, nullptr, HB,
             FFND, SEQ, FFND, HIDN, 1);
        gemm(HB, FFND, W2T + (size_t)l * HIDN * FFND, FFND, b2 + l * HIDN, OF, nullptr,
             HIDN, SEQ, HIDN, FFND, 0);
        resid_ln_kernel<<<SEQ, 256, 0, stream>>>(XF, OF, ln2_s + l * HIDN,
                                                 ln2_b + l * HIDN, XB);
    }

    // final vocab projection (f32 output)
    gemm(XB, HIDN, WfcT, HIDN, bfc, (float*)d_out, nullptr, VOCAB, SEQ, VOCAB, HIDN, 0);
}
